// MultiHeadAttention_28432683499936
// MI455X (gfx1250) — matrix operand order repaired and checked by reading, not compiled, not run on hardware
//
#include <hip/hip_runtime.h>
#include <hip/hip_bf16.h>

typedef __bf16 bf16;
typedef __attribute__((ext_vector_type(16))) __bf16 v16bf;
typedef __attribute__((ext_vector_type(8)))  __bf16 bf16x8;
typedef __attribute__((ext_vector_type(8)))  float  v8f;

#define NEGINF  (-3.0e38f)
// (1/sqrt(1024)) * log2(e): fold softmax scale into exp2 domain
#define SCALE2  (0.03125f * 1.44269504088896340736f)

__device__ __forceinline__ v8f wmma_bf16(v16bf a, v16bf b, v8f c) {
  // D(f32 16x16) = A(16x32 bf16) * B(32x16 bf16) + C
  return __builtin_amdgcn_wmma_f32_16x16x32_bf16(false, a, false, b, (short)0, c, false, false);
}

union FragU { v16bf v; bf16x8 h[2]; };

// A-matrix 16x32 (16-bit): lane m=L&15 holds contiguous K chunk [hi*16,+16)
__device__ __forceinline__ v16bf load_a_frag(const bf16* __restrict__ row, int hi) {
  FragU u;
  u.h[0] = *(const bf16x8*)(row + hi * 16);
  u.h[1] = *(const bf16x8*)(row + hi * 16 + 8);
  return u.v;
}

// B-matrix 32x16 (16-bit): lane n=L&15 holds contiguous K chunk [hi*16,+16)
__device__ __forceinline__ v16bf load_b_frag(const bf16* __restrict__ row, int hi) {
  FragU u;
  const bf16* p = row + hi * 16;
  u.h[0] = *(const bf16x8*)(p);
  u.h[1] = *(const bf16x8*)(p + 8);
  return u.v;
}

// --------------------------------------------------------------------------
// Async global->LDS staging (CDNA5 ASYNCcnt path), with safe fallback.
// Builtin signature (from hipcc diagnostic): (int4 AS1*, int4 AS3*, Ii, Ii).
// --------------------------------------------------------------------------
#if defined(__has_builtin)
#if __has_builtin(__builtin_amdgcn_global_load_async_to_lds_b128) && \
    __has_builtin(__builtin_amdgcn_s_wait_asynccnt)
#define HAVE_ASYNC_LDS 1
#endif
#endif

typedef __attribute__((ext_vector_type(4))) int v4i;
typedef __attribute__((address_space(1))) v4i v4i_g;   // global int4
typedef __attribute__((address_space(3))) v4i v4i_l;   // LDS int4

__device__ __forceinline__ void async_cp16(void* lp, const void* gp) {
#ifdef HAVE_ASYNC_LDS
  __builtin_amdgcn_global_load_async_to_lds_b128((v4i_g*)gp, (v4i_l*)lp, 0, 0);
#else
  *(uint4*)lp = *(const uint4*)gp;  // synchronous fallback
#endif
}

__device__ __forceinline__ void async_wait_le(int pending) {
#ifdef HAVE_ASYNC_LDS
  if (pending) __builtin_amdgcn_s_wait_asynccnt(4);
  else         __builtin_amdgcn_s_wait_asynccnt(0);
#else
  (void)pending;
#endif
}

// ---------------------------------------------------------------------------
// f32 -> bf16 conversion
// ---------------------------------------------------------------------------
__global__ void cvt_f32_bf16(const float* __restrict__ in, bf16* __restrict__ out, int nElem) {
  int i = blockIdx.x * blockDim.x + threadIdx.x;
  if (i < nElem) out[i] = (bf16)in[i];
}

// ---------------------------------------------------------------------------
// Pack W[1024][1024] (row n = output col, y = x*W^T) into the exact per-lane
// WMMA B-fragment layout:  Wp[((nt*32 + kt)*32 + L)*16 + e]
//   nt: 16-col tile (0..63), kt: 32-k tile (0..31), L: lane, e: 0..15
//   lane L holds col n = nt*16 + (L&15), k = kt*32 + (L>>4)*16 + e
// => a wave fetches one whole B fragment as a contiguous, coalesced 1KB block.
// ---------------------------------------------------------------------------
__global__ void cvt_pack_w(const float* __restrict__ W, bf16* __restrict__ Wp, int nElem) {
  int i = blockIdx.x * blockDim.x + threadIdx.x;
  if (i >= nElem) return;
  int e  = i & 15;
  int L  = (i >> 4) & 31;
  int kt = (i >> 9) & 31;
  int nt = i >> 14;
  int n  = nt * 16 + (L & 15);
  int k  = kt * 32 + (L >> 4) * 16 + e;
  Wp[i] = (bf16)W[(size_t)n * 1024 + k];
}

__device__ __forceinline__ v16bf load_b_packed(const bf16* __restrict__ Wp,
                                               int nt, int kt, int lane) {
  return *(const v16bf*)(Wp + (((size_t)nt * 32 + kt) * 32 + lane) * 16);
}

// ---------------------------------------------------------------------------
// GEMM: out[m][n] = sum_k A[m][k] * W[n][k]; A:[4096][1024] bf16 (row major),
// Wp packed (see above). Wave tile 32x64: 2 A-frags x 4 B-frags -> 8 WMMA/step.
// mode 0: bf16 store [4096][1024]
// mode 1: bf16 store transposed per head -> Vt[b][h][d][t] (t fastest, 2048)
// mode 2: f32 store [4096][1024] (final output)
// ---------------------------------------------------------------------------
__global__ __launch_bounds__(128) void gemm_bf16_wmma(
    const bf16* __restrict__ A, const bf16* __restrict__ Wp,
    void* __restrict__ out, int mode)
{
  const int lane = threadIdx.x & 31;
  const int wv   = threadIdx.x >> 5;
  const int n    = lane & 15;
  const int hi   = lane >> 4;
  const int mg   = blockIdx.x;                   // 128 groups of 32 rows
  const int wng  = blockIdx.y * 4 + wv;          // 16 n-groups of 64 cols
  const int n0   = wng * 64;

  const bf16* arow0 = A + (size_t)(mg * 32 + n) * 1024;       // m-tile 0
  const bf16* arow1 = arow0 + (size_t)16 * 1024;              // m-tile 1
  v8f acc[2][4] = {};

  for (int kt = 0; kt < 32; ++kt) {
    const int kk = kt * 32;
    v16bf a0 = load_a_frag(arow0 + kk, hi);
    v16bf a1 = load_a_frag(arow1 + kk, hi);
#pragma unroll
    for (int j = 0; j < 4; ++j) {
      v16bf b = load_b_packed(Wp, wng * 4 + j, kt, lane);
      acc[0][j] = wmma_bf16(a0, b, acc[0][j]);
      acc[1][j] = wmma_bf16(a1, b, acc[1][j]);
    }
  }

#pragma unroll
  for (int mi = 0; mi < 2; ++mi) {
    const int m0 = mg * 32 + mi * 16;
    if (mode == 0) {
      bf16* O = (bf16*)out;
#pragma unroll
      for (int j = 0; j < 4; ++j)
#pragma unroll
        for (int r = 0; r < 8; ++r)
          O[(size_t)(m0 + r + 8 * hi) * 1024 + n0 + j * 16 + n] = (bf16)acc[mi][j][r];
    } else if (mode == 1) {
      bf16* Vt = (bf16*)out;
      const int b  = m0 >> 11;             // token rows are b*2048 + t
      const int t0 = (m0 & 2047) + 8 * hi;
#pragma unroll
      for (int j = 0; j < 4; ++j) {
        int ng = n0 + j * 16 + n;          // global col = h*64 + d
        int h = ng >> 6, d = ng & 63;
        bf16x8 pk;
#pragma unroll
        for (int r = 0; r < 8; ++r) pk[r] = (bf16)acc[mi][j][r];
        *(bf16x8*)(Vt + ((size_t)((b * 16 + h) * 64 + d)) * 2048 + t0) = pk;
      }
    } else {
      float* O = (float*)out;
#pragma unroll
      for (int j = 0; j < 4; ++j)
#pragma unroll
        for (int r = 0; r < 8; ++r)
          O[(size_t)(m0 + r + 8 * hi) * 1024 + n0 + j * 16 + n] = acc[mi][j][r];
    }
  }
}

// ---------------------------------------------------------------------------
// Causal flash attention. Block = 4 waves sharing (b,h), wave = one 16-row
// q-tile, 32 keys per step. K/V tiles are cooperatively staged into LDS with
// async global->LDS copies (double buffered); waves read WMMA B-fragments
// from LDS with conflict-free padded strides.
// Q,K: bf16 [b*2048][1024]; Vt: bf16 [b][h][64][2048]; O: bf16 [b*2048][1024]
// ---------------------------------------------------------------------------
__global__ __launch_bounds__(128) void flash_attn_wmma(
    const bf16* __restrict__ Q, const bf16* __restrict__ K,
    const bf16* __restrict__ Vt, bf16* __restrict__ O)
{
  __shared__ __align__(16) bf16 Ktile[2][32][72];  // [buf][key][d],  72*2B = 36-bank stride
  __shared__ __align__(16) bf16 Vtile[2][64][40];  // [buf][d][key],  40*2B = 20-bank stride
  __shared__ __align__(16) bf16 plds[4][16][40];   // per-wave P tile

  const int tid  = threadIdx.x;
  const int lane = tid & 31;
  const int wv   = tid >> 5;
  const int n    = lane & 15;
  const int hi   = lane >> 4;
  const int bh   = blockIdx.x;            // 0..31  (b*16 + h)
  const int b    = bh >> 4;
  const int h    = bh & 15;
  const int qg   = blockIdx.y;            // 0..31 q-tile group (4 tiles)
  const int qt   = qg * 4 + wv;
  const int q0   = qt * 16;
  const int bbase = b * 2048;
  const int dbase = h * 64;

  const bf16* Kg0 = K  + (size_t)bbase * 1024 + dbase;   // + (kk0+row)*1024 + col
  const bf16* Vg0 = Vt + (size_t)(bh * 64) * 2048;       // + row*2048 + kk0 + col

  // Q fragments for d in [0,32) and [32,64), reused across all key steps
  const bf16* qrow = Q + (size_t)(bbase + q0 + n) * 1024 + dbase;
  const v16bf aq0 = load_a_frag(qrow, hi);
  const v16bf aq1 = load_a_frag(qrow + 32, hi);

  v8f acc[4] = {};
  float mrow[8], lrow[8];
#pragma unroll
  for (int r = 0; r < 8; ++r) { mrow[r] = NEGINF; lrow[r] = 0.0f; }

  const int mySteps = (qt >> 1) + 1;      // this wave's causal step count
  const int smax    = 2 * qg + 2;         // block-uniform max (wave wv=3)

  // ---- stage step 0 ----
  {
#pragma unroll
    for (int rr = 0; rr < 2; ++rr) {
      int flat = tid * 8 + rr * 1024;                 // 32x64 K tile
      int row = flat >> 6, col = flat & 63;
      async_cp16(&Ktile[0][row][col], Kg0 + (size_t)row * 1024 + col);
    }
#pragma unroll
    for (int rr = 0; rr < 2; ++rr) {
      int flat = tid * 8 + rr * 1024;                 // 64x32 V tile
      int row = flat >> 5, col = flat & 31;
      async_cp16(&Vtile[0][row][col], Vg0 + (size_t)row * 2048 + col);
    }
  }

  for (int s = 0; s < smax; ++s) {
    const int buf = s & 1;
    const int nxt = s + 1;
    if (nxt < smax) {                      // prefetch next tile into other buffer
      const int kk1 = nxt * 32;
#pragma unroll
      for (int rr = 0; rr < 2; ++rr) {
        int flat = tid * 8 + rr * 1024;
        int row = flat >> 6, col = flat & 63;
        async_cp16(&Ktile[nxt & 1][row][col], Kg0 + (size_t)(kk1 + row) * 1024 + col);
      }
#pragma unroll
      for (int rr = 0; rr < 2; ++rr) {
        int flat = tid * 8 + rr * 1024;
        int row = flat >> 5, col = flat & 31;
        async_cp16(&Vtile[nxt & 1][row][col], Vg0 + (size_t)row * 2048 + kk1 + col);
      }
    }
    async_wait_le(nxt < smax);             // drain step-s copies (in-order done)
    __syncthreads();                       // tile visible to all waves

    if (s < mySteps) {
      const int kk0 = s * 32;

      // S = Q * K^T for two 16-key tiles (d=64 -> 2 chained WMMAs each)
      v16bf b00 = load_b_frag(&Ktile[buf][n][0],       hi);
      v16bf b01 = load_b_frag(&Ktile[buf][n][32],      hi);
      v16bf b10 = load_b_frag(&Ktile[buf][16 + n][0],  hi);
      v16bf b11 = load_b_frag(&Ktile[buf][16 + n][32], hi);

      v8f s0 = {}, s1 = {};
      s0 = wmma_bf16(aq0, b00, s0);
      s0 = wmma_bf16(aq1, b01, s0);
      s1 = wmma_bf16(aq0, b10, s1);
      s1 = wmma_bf16(aq1, b11, s1);

      const bool diag = (kk0 + 31) > q0;   // only last causal step masks
      float p0[8], p1[8];
#pragma unroll
      for (int r = 0; r < 8; ++r) {
        float x0 = s0[r] * SCALE2;
        float x1 = s1[r] * SCALE2;
        if (diag) {
          int q = q0 + r + 8 * hi;
          if (kk0 + n      > q) x0 = NEGINF;
          if (kk0 + 16 + n > q) x1 = NEGINF;
        }
        float mx = fmaxf(x0, x1);
        mx = fmaxf(mx, __shfl_xor(mx, 1, 32));
        mx = fmaxf(mx, __shfl_xor(mx, 2, 32));
        mx = fmaxf(mx, __shfl_xor(mx, 4, 32));
        mx = fmaxf(mx, __shfl_xor(mx, 8, 32));
        float mnew = fmaxf(mrow[r], mx);
        float e0 = exp2f(x0 - mnew);
        float e1 = exp2f(x1 - mnew);
        float rs = e0 + e1;
        rs += __shfl_xor(rs, 1, 32);
        rs += __shfl_xor(rs, 2, 32);
        rs += __shfl_xor(rs, 4, 32);
        rs += __shfl_xor(rs, 8, 32);
        float f = exp2f(mrow[r] - mnew);   // first iter: exp2(-3e38-m) == 0
        lrow[r] = lrow[r] * f + rs;
        mrow[r] = mnew;
        p0[r] = e0;
        p1[r] = e1;
#pragma unroll
        for (int j = 0; j < 4; ++j) acc[j][r] *= f;
      }

      // P (C/D layout) -> per-wave LDS tile -> A-layout fragment.
      // Same-wave DS ops are in-order (ISA 7.3); only compile-time ordering needed.
#pragma unroll
      for (int r = 0; r < 8; ++r) {
        plds[wv][r + 8 * hi][n]      = (bf16)p0[r];
        plds[wv][r + 8 * hi][n + 16] = (bf16)p1[r];
      }
      __builtin_amdgcn_wave_barrier();
      FragU up;
      up.h[0] = *(const bf16x8*)(&plds[wv][n][hi * 16]);
      up.h[1] = *(const bf16x8*)(&plds[wv][n][hi * 16 + 8]);
      v16bf ap = up.v;
      __builtin_amdgcn_wave_barrier();

      // O += P * V from the staged (pre-transposed) V tile
#pragma unroll
      for (int j = 0; j < 4; ++j) {
        v16bf bv = load_b_frag(&Vtile[buf][j * 16 + n][0], hi);
        acc[j] = wmma_bf16(ap, bv, acc[j]);
      }
    }
    __syncthreads();                       // everyone done with buf before reuse
  }

  // normalize by row sums and store O in [b*2048][1024] bf16 layout
#pragma unroll
  for (int r = 0; r < 8; ++r) {
    float inv = 1.0f / lrow[r];
#pragma unroll
    for (int j = 0; j < 4; ++j) {
      float o = acc[j][r] * inv;
      O[(size_t)(bbase + q0 + r + 8 * hi) * 1024 + dbase + j * 16 + n] = (bf16)o;
    }
  }
}

// ---------------------------------------------------------------------------
// Host-side launch
// ---------------------------------------------------------------------------
extern "C" void kernel_launch(void* const* d_in, const int* in_sizes, int n_in,
                              void* d_out, int out_size, void* d_ws, size_t ws_size,
                              hipStream_t stream) {
  (void)in_sizes; (void)n_in; (void)out_size; (void)ws_size;
  const float* x  = (const float*)d_in[0];
  const float* Wq = (const float*)d_in[1];
  const float* Wk = (const float*)d_in[2];
  const float* Wv = (const float*)d_in[3];
  const float* Wo = (const float*)d_in[4];

  const size_t NX = (size_t)4096 * 1024;   // (b*t) x k
  const size_t NW = (size_t)1024 * 1024;

  char* p = (char*)d_ws;
  bf16* xb  = (bf16*)p; p += NX * sizeof(bf16);   //  8 MB
  bf16* Wqp = (bf16*)p; p += NW * sizeof(bf16);   //  2 MB (packed frag layout)
  bf16* Wkp = (bf16*)p; p += NW * sizeof(bf16);
  bf16* Wvp = (bf16*)p; p += NW * sizeof(bf16);
  bf16* Wop = (bf16*)p; p += NW * sizeof(bf16);
  bf16* Qb  = (bf16*)p; p += NX * sizeof(bf16);   //  8 MB
  bf16* Kb  = (bf16*)p; p += NX * sizeof(bf16);
  bf16* Vt  = (bf16*)p; p += NX * sizeof(bf16);   // [b][h][64][2048]
  bf16* Ob  = (bf16*)p; p += NX * sizeof(bf16);   // total 48 MB

  cvt_f32_bf16<<<(int)((NX + 255) / 256), 256, 0, stream>>>(x, xb, (int)NX);
  cvt_pack_w <<<(int)((NW + 255) / 256), 256, 0, stream>>>(Wq, Wqp, (int)NW);
  cvt_pack_w <<<(int)((NW + 255) / 256), 256, 0, stream>>>(Wk, Wkp, (int)NW);
  cvt_pack_w <<<(int)((NW + 255) / 256), 256, 0, stream>>>(Wv, Wvp, (int)NW);
  cvt_pack_w <<<(int)((NW + 255) / 256), 256, 0, stream>>>(Wo, Wop, (int)NW);

  dim3 gg(128, 4), gb(128, 1);
  gemm_bf16_wmma<<<gg, gb, 0, stream>>>(xb, Wqp, (void*)Qb, 0);
  gemm_bf16_wmma<<<gg, gb, 0, stream>>>(xb, Wkp, (void*)Kb, 0);
  gemm_bf16_wmma<<<gg, gb, 0, stream>>>(xb, Wvp, (void*)Vt, 1);

  flash_attn_wmma<<<dim3(32, 32), 128, 0, stream>>>(Qb, Kb, Vt, Ob);

  gemm_bf16_wmma<<<gg, gb, 0, stream>>>(Ob, Wop, d_out, 2);
}
